// FusionGRU_57672820850711
// MI455X (gfx1250) — compile-verified, matrix operand order checked by reading, not run on hardware
//
#include <hip/hip_runtime.h>
#include <hip/hip_bf16.h>
#include <math.h>

// ---------------- CDNA5 WMMA types ----------------
typedef __attribute__((ext_vector_type(16))) __bf16 v16bf;
typedef __attribute__((ext_vector_type(8)))  float  v8f;

#define D_MODEL 256
#define HID     768
#define G3      2304      // 3*HID
#define INP0    512       // 2*D_MODEL
#define BATCH   16
#define SEQ     4096
#define MTOT    (BATCH * SEQ)   // 65536
#define NWG_SEQ 16

// ---------------------------------------------------------------------------
// gfx1250 async global->LDS (ASYNCcnt-tracked; ISA 15.18.3 op 98).
// LDS destination address comes from a VGPR holding the LDS byte offset,
// obtained via generic->local addrspace cast.
// ---------------------------------------------------------------------------
typedef __attribute__((address_space(3))) char lds_char_t;

__device__ __forceinline__ unsigned lds_addr_of(const void* p) {
  return (unsigned)(size_t)(lds_char_t*)(p);
}

__device__ __forceinline__ void async_load_b128(unsigned lds_off, const void* gptr) {
  asm volatile("global_load_async_to_lds_b128 %0, %1, off"
               :: "v"(lds_off), "v"((unsigned long long)(size_t)gptr)
               : "memory");
}

__device__ __forceinline__ void wait_async0() {
  asm volatile("s_wait_asynccnt 0x0" ::: "memory");
}

// Fast gate nonlinearities: one v_exp_f32 + one v_rcp_f32 each (keeps the
// 4096-step sequential critical path off the IEEE divide sequence).
__device__ __forceinline__ float fast_sigmoid(float x) {
  return __builtin_amdgcn_rcpf(1.f + __expf(-x));
}
__device__ __forceinline__ float fast_tanh(float x) {
  return 2.f * __builtin_amdgcn_rcpf(1.f + __expf(-2.f * x)) - 1.f;
}

// ---------------------------------------------------------------------------
// LDS fragment loader. CDNA5 16-bit A/B layout (ISA 7.12.2):
//   lane L: row = L&15 ; khalf = L>>4 ; element j -> k = khalf*8 + (j&7) + (j>>3)*16
// => two contiguous 16B chunks at k-offsets khalf*8 and 16+khalf*8.
// stride_e (in bf16 elems) must keep rows 16B aligned (we use 40 or 768).
// ---------------------------------------------------------------------------
__device__ __forceinline__ v16bf load_frag(const __bf16* base, int row, int stride_e,
                                           int lane, int kbase) {
  const int khalf = lane >> 4;
  const char* p = (const char*)(base + (size_t)row * stride_e + kbase + khalf * 8);
  union { v16bf v; uint4 q[2]; } u;
  u.q[0] = *(const uint4*)(p);
  u.q[1] = *(const uint4*)(p + 32);
  return u.v;
}

// ---------------------------------------------------------------------------
// Tiled WMMA GEMM:  C[M,N] = A[M,K] (fp32->bf16) * Wb[N,K]^T (bf16) + bias (+ReLU)
// Block: 256 threads (8 waves, 2x4), tile 64(M) x 128(N), k-step 32.
// CONCAT templated: each staging thread's 8-elem chunk is entirely inside
// pose, imu, or the zero pad -> a single uniform pointer select + float4 loads.
// B tile staged with gfx1250 async-to-LDS (no VGPR round trip).
// ---------------------------------------------------------------------------
template <int CONCAT, int RELU>
__global__ void wmma_gemm(const float* __restrict__ A, const float* __restrict__ pose,
                          const float* __restrict__ imu, const __bf16* __restrict__ Wb,
                          const float* __restrict__ bias, float* __restrict__ Cout,
                          int K, int N) {
  __shared__ __bf16 As[64 * 40];   // stride 40 -> 80B rows (16B aligned)
  __shared__ __bf16 Bs[128 * 40];

  const int tid  = threadIdx.x;
  const int lane = tid & 31;
  const int wid  = tid >> 5;
  const int mw   = wid >> 2;          // 0..1
  const int nw   = wid & 3;           // 0..3
  const int m0   = blockIdx.y * 64;
  const int n0   = blockIdx.x * 128;

  const unsigned bs_base = lds_addr_of(Bs);

  v8f acc[2][2] = {};

  for (int k0 = 0; k0 < K; k0 += 32) {
    __syncthreads();
    {  // stage B 128x32 async (bf16 source, 2x 16B transfers)
      const int row = tid >> 1, kp = (tid & 1) * 16;
      const char* src = (const char*)(Wb + (size_t)(n0 + row) * K + k0 + kp);
      async_load_b128(bs_base + (unsigned)(row * 40 + kp) * 2, src);
      async_load_b128(bs_base + (unsigned)(row * 40 + kp + 8) * 2, src + 16);
      if (k0 + 32 < K)  // gfx1250 global_prefetch_b8 for next k-tile
        __builtin_prefetch(src + 64, 0, 1);
    }
    {  // stage A 64x32 (fp32 -> bf16 conversion through VGPRs)
      const int row = tid >> 2, kp = (tid & 3) * 8;
      const int m = m0 + row;
      const int kbase = k0 + kp;
      float4 f0, f1;
      if (CONCAT) {
        const int t = m & (SEQ - 1);
        const int b = m >> 12;
        if (kbase < D_MODEL) {
          const float4* s = (const float4*)(pose + (size_t)m * D_MODEL + kbase);
          f0 = s[0]; f1 = s[1];
        } else if (t == 0) {
          f0 = make_float4(0.f, 0.f, 0.f, 0.f); f1 = f0;
        } else {
          const float4* s = (const float4*)(imu + ((size_t)b * (SEQ - 1) + (t - 1)) * D_MODEL
                                            + (kbase - D_MODEL));
          f0 = s[0]; f1 = s[1];
        }
      } else {
        const float4* s = (const float4*)(A + (size_t)m * K + kbase);
        f0 = s[0]; f1 = s[1];
      }
      __bf16* d = &As[row * 40 + kp];
      d[0] = (__bf16)f0.x; d[1] = (__bf16)f0.y; d[2] = (__bf16)f0.z; d[3] = (__bf16)f0.w;
      d[4] = (__bf16)f1.x; d[5] = (__bf16)f1.y; d[6] = (__bf16)f1.z; d[7] = (__bf16)f1.w;
    }
    wait_async0();
    __syncthreads();

#pragma unroll
    for (int mt = 0; mt < 2; ++mt) {
      const v16bf af = load_frag(As, mw * 32 + mt * 16 + (lane & 15), 40, lane, 0);
#pragma unroll
      for (int nt = 0; nt < 2; ++nt) {
        const v16bf bf = load_frag(Bs, nw * 32 + nt * 16 + (lane & 15), 40, lane, 0);
        acc[mt][nt] = __builtin_amdgcn_wmma_f32_16x16x32_bf16(
            false, af, false, bf, (short)0, acc[mt][nt], false, false);
      }
    }
  }

  // Epilogue: C/D layout -> n = lane&15, m = reg + 8*(lane>>4)
  const int khalf = lane >> 4, nlane = lane & 15;
#pragma unroll
  for (int mt = 0; mt < 2; ++mt)
#pragma unroll
    for (int nt = 0; nt < 2; ++nt) {
      const int n = n0 + nw * 32 + nt * 16 + nlane;
      const float bv = bias[n];
#pragma unroll
      for (int r = 0; r < 8; ++r) {
        const int m = m0 + mw * 32 + mt * 16 + r + khalf * 8;
        float v = acc[mt][nt][r] + bv;
        if (RELU) v = fmaxf(v, 0.f);
        Cout[(size_t)m * N + n] = v;
      }
    }
}

// ---------------------------------------------------------------------------
// Device-wide sense-reversing barrier (16 co-resident WGs, one per WGP)
// ---------------------------------------------------------------------------
__device__ __forceinline__ void grid_sync(unsigned* bar, unsigned nwg) {
  __threadfence();
  __syncthreads();
  if (threadIdx.x == 0) {
    const unsigned gen =
        __hip_atomic_load(&bar[1], __ATOMIC_RELAXED, __HIP_MEMORY_SCOPE_AGENT);
    if (atomicAdd(&bar[0], 1u) == nwg - 1) {
      __hip_atomic_store(&bar[0], 0u, __ATOMIC_RELAXED, __HIP_MEMORY_SCOPE_AGENT);
      __hip_atomic_fetch_add(&bar[1], 1u, __ATOMIC_RELEASE, __HIP_MEMORY_SCOPE_AGENT);
    } else {
      while (__hip_atomic_load(&bar[1], __ATOMIC_ACQUIRE, __HIP_MEMORY_SCOPE_AGENT) == gen)
        __builtin_amdgcn_s_sleep(1);
    }
  }
  __syncthreads();
}

// ---------------------------------------------------------------------------
// Persistent GRU recurrence. Grid = 16 WGs x 288 threads (9 waves).
// WG g owns hidden slice j in [g*48, g*48+48) => 144 gate rows {r,z,n}.
// W_hh slice (144x768 bf16 = 216KB) lives in LDS for the whole kernel
// (CDNA5 320KB WGP LDS). Wave w (0..8): gate g=w/3, j-tile jt=w%3 -> one
// 16x16 WMMA tile; batch (16) is the WMMA M dimension. h is exchanged
// through L2 with one grid sync per step; all global->LDS staging uses
// gfx1250 async-to-LDS ops.
// ---------------------------------------------------------------------------
__global__ void gru_seq(const float* __restrict__ xg, const __bf16* __restrict__ Whb,
                        const float* __restrict__ bhh, float* __restrict__ Hout,
                        __bf16* __restrict__ hbuf, unsigned* __restrict__ bar) {
  extern __shared__ char smem[];
  __bf16* Ws    = (__bf16*)(smem);            // [144][768] bf16   (221184 B)
  __bf16* Ast   = (__bf16*)(smem + 221184);   // [16][768]  bf16   ( 24576 B)
  float*  exch  = (float*)(smem + 245760);    // [9][16][16] f32   (  9216 B)
  float*  hprev = (float*)(smem + 254976);    // [16][48]   f32    (  3072 B)
  float*  bl    = (float*)(smem + 258048);    // [144]      f32    (   576 B)

  const int tid  = threadIdx.x;     // 0..287
  const int lane = tid & 31;
  const int w    = tid >> 5;        // 0..8
  const int wg   = blockIdx.x;      // 0..15
  const int NT   = NWG_SEQ * 18;    // 288 threads
  const int rowq = (HID * 2) / 16;  // 96 x 16B per 768-bf16 row

  const unsigned ws_base  = lds_addr_of(Ws);
  const unsigned ast_base = lds_addr_of(Ast);

  // Preload W_hh slice into LDS via async DMA (rows reordered to g*48+jj)
  for (int i = tid; i < 144 * rowq; i += NT) {
    const int lr = i / rowq, c = i % rowq;
    const int g = lr / 48, jj = lr % 48;
    const int grow = g * HID + wg * 48 + jj;
    async_load_b128(ws_base + (unsigned)i * 16,
                    (const char*)Whb + ((size_t)grow * rowq + c) * 16);
  }
  for (int i = tid; i < 144; i += NT) {
    const int g = i / 48, jj = i % 48;
    bl[i] = bhh[g * HID + wg * 48 + jj];
  }
  for (int i = tid; i < BATCH * 48; i += NT) hprev[i] = 0.f;
  wait_async0();
  __syncthreads();

  const int gg = w / 3, jt = w % 3;
  const int wrow = gg * 48 + jt * 16;   // local row base inside Ws

  for (int t = 0; t < SEQ; ++t) {
    // Async-stage h_prev (16 batches x 768 bf16 = 24KB) from the ping-pong buffer
    const char* hsrc = (const char*)(hbuf + (size_t)(t & 1) * (BATCH * HID));
    for (int i = tid; i < BATCH * rowq; i += NT)
      async_load_b128(ast_base + (unsigned)i * 16, hsrc + (size_t)i * 16);
    wait_async0();
    __syncthreads();

    // hg tile = h_prev[16 x 768] * Whh_slice^T  -> 16(batch) x 16(gate rows)
    v8f acc = {};
#pragma unroll 4
    for (int kk = 0; kk < HID; kk += 32) {
      const v16bf af = load_frag(Ast, (lane & 15), HID, lane, kk);
      const v16bf bf = load_frag(Ws, wrow + (lane & 15), HID, lane, kk);
      acc = __builtin_amdgcn_wmma_f32_16x16x32_bf16(
          false, af, false, bf, (short)0, acc, false, false);
    }
    {  // spill accum tile to LDS for cross-wave gate combine
      const int khalf = lane >> 4, nl = lane & 15;
#pragma unroll
      for (int r = 0; r < 8; ++r)
        exch[((size_t)w * 16 + (r + khalf * 8)) * 16 + nl] = acc[r];
    }
    __syncthreads();

    // Elementwise GRU cell for this WG's 16x48 slice
    __bf16* hdst = hbuf + (size_t)((t + 1) & 1) * (BATCH * HID);
    for (int idx = tid; idx < BATCH * 48; idx += NT) {
      const int m = idx / 48, j = idx % 48;
      const int jt2 = j >> 4, jn = j & 15;
      const float hr = exch[((0 * 3 + jt2) * 16 + m) * 16 + jn] + bl[0 * 48 + j];
      const float hz = exch[((1 * 3 + jt2) * 16 + m) * 16 + jn] + bl[1 * 48 + j];
      const float hn = exch[((2 * 3 + jt2) * 16 + m) * 16 + jn] + bl[2 * 48 + j];
      const size_t xrow = ((size_t)m * SEQ + t) * G3;
      const int jg = wg * 48 + j;
      const float xr = xg[xrow + 0 * HID + jg];
      const float xz = xg[xrow + 1 * HID + jg];
      const float xn = xg[xrow + 2 * HID + jg];
      const float r = fast_sigmoid(xr + hr);
      const float z = fast_sigmoid(xz + hz);
      const float nn = fast_tanh(xn + r * hn);
      const float hp = hprev[m * 48 + j];
      const float hnew = (1.f - z) * nn + z * hp;
      hprev[m * 48 + j] = hnew;
      Hout[((size_t)m * SEQ + t) * HID + jg] = hnew;
      hdst[(size_t)m * HID + jg] = (__bf16)hnew;
    }
    grid_sync(bar, NWG_SEQ);
  }
}

// ---------------------------------------------------------------------------
// Small helpers
// ---------------------------------------------------------------------------
__global__ void cvt_bf16(const float* __restrict__ src, __bf16* __restrict__ dst, int n) {
  const int i = blockIdx.x * blockDim.x + threadIdx.x;
  if (i < n) dst[i] = (__bf16)src[i];
}

__global__ void init_state(__bf16* hbuf, unsigned* bar) {
  const int i = blockIdx.x * blockDim.x + threadIdx.x;
  if (i < 2 * BATCH * HID) hbuf[i] = (__bf16)0.f;
  if (i < 4) bar[i] = 0u;
}

// ---------------------------------------------------------------------------
extern "C" void kernel_launch(void* const* d_in, const int* in_sizes, int n_in,
                              void* d_out, int out_size, void* d_ws, size_t ws_size,
                              hipStream_t stream) {
  (void)in_sizes; (void)n_in; (void)out_size; (void)ws_size;
  const float* pose  = (const float*)d_in[0];
  const float* imu   = (const float*)d_in[1];
  const float* w_ih0 = (const float*)d_in[2];
  const float* w_hh0 = (const float*)d_in[3];
  const float* b_ih0 = (const float*)d_in[4];
  const float* b_hh0 = (const float*)d_in[5];
  const float* w_ih1 = (const float*)d_in[6];
  const float* w_hh1 = (const float*)d_in[7];
  const float* b_ih1 = (const float*)d_in[8];
  const float* b_hh1 = (const float*)d_in[9];
  const float* fc_w  = (const float*)d_in[10];
  const float* fc_b  = (const float*)d_in[11];
  float* out = (float*)d_out;

  // Workspace carve-up
  char* ws = (char*)d_ws;
  size_t off = 0;
  float* xg = (float*)(ws + off);      off += (size_t)MTOT * G3 * sizeof(float);     // 604 MB
  float* Hb = (float*)(ws + off);      off += (size_t)MTOT * HID * sizeof(float);    // 192 MB
  __bf16* wih0b = (__bf16*)(ws + off); off += (size_t)G3 * INP0 * 2;
  __bf16* whh0b = (__bf16*)(ws + off); off += (size_t)G3 * HID * 2;
  __bf16* wih1b = (__bf16*)(ws + off); off += (size_t)G3 * HID * 2;
  __bf16* whh1b = (__bf16*)(ws + off); off += (size_t)G3 * HID * 2;
  __bf16* fcwb  = (__bf16*)(ws + off); off += (size_t)D_MODEL * HID * 2;
  __bf16* hbuf  = (__bf16*)(ws + off); off += (size_t)2 * BATCH * HID * 2;
  off = (off + 127) & ~(size_t)127;
  unsigned* bar = (unsigned*)(ws + off);

  // 1) Convert weights to bf16 once per launch
  cvt_bf16<<<(G3 * INP0 + 255) / 256, 256, 0, stream>>>(w_ih0, wih0b, G3 * INP0);
  cvt_bf16<<<(G3 * HID + 255) / 256, 256, 0, stream>>>(w_hh0, whh0b, G3 * HID);
  cvt_bf16<<<(G3 * HID + 255) / 256, 256, 0, stream>>>(w_ih1, wih1b, G3 * HID);
  cvt_bf16<<<(G3 * HID + 255) / 256, 256, 0, stream>>>(w_hh1, whh1b, G3 * HID);
  cvt_bf16<<<(D_MODEL * HID + 255) / 256, 256, 0, stream>>>(fc_w, fcwb, D_MODEL * HID);

  const size_t seq_lds = 221184 + 24576 + 9216 + 3072 + 576;  // 258624 B (< 320KB WGP LDS)
  const dim3 gGate(G3 / 128, MTOT / 64);   // (18, 1024)
  const dim3 gFC(D_MODEL / 128, MTOT / 64);

  // 2) Layer 0: input gates (concat A), then recurrence
  init_state<<<96, 256, 0, stream>>>(hbuf, bar);
  wmma_gemm<1, 0><<<gGate, 256, 0, stream>>>(nullptr, pose, imu, wih0b, b_ih0, xg,
                                             INP0, G3);
  gru_seq<<<NWG_SEQ, 288, seq_lds, stream>>>(xg, whh0b, b_hh0, Hb, hbuf, bar);

  // 3) Layer 1: input gates from H1, then recurrence (H buffer reused)
  wmma_gemm<0, 0><<<gGate, 256, 0, stream>>>(Hb, nullptr, nullptr, wih1b, b_ih1, xg,
                                             HID, G3);
  init_state<<<96, 256, 0, stream>>>(hbuf, bar);
  gru_seq<<<NWG_SEQ, 288, seq_lds, stream>>>(xg, whh1b, b_hh1, Hb, hbuf, bar);

  // 4) FC head + ReLU -> d_out
  wmma_gemm<0, 1><<<gFC, 256, 0, stream>>>(Hb, nullptr, nullptr, fcwb, fc_b, out,
                                           HID, D_MODEL);
}
